// CRF_66743791780267
// MI455X (gfx1250) — compile-verified, matrix operand order checked by reading, not run on hardware
//
#include <hip/hip_runtime.h>
#include <cstddef>

// ---------------------------------------------------------------------------
// Dense CRF (96x96, 21 labels) for MI455X (gfx1250, wave32, WMMA).
//   - K matrix (9216x9216) materialized once in bf16 (170 MB, L2-resident)
//   - bilateral term = bf16 WMMA GEMM [32 x 9216] x [9216 x 9216], split-K
//   - spatial term = separable 71-tap Gaussian
//   - softmax fused in combine kernel
// ---------------------------------------------------------------------------

typedef __bf16 bf16;
typedef __bf16 v16bf __attribute__((ext_vector_type(16)));
typedef __bf16 v8bf  __attribute__((ext_vector_type(8)));
typedef float  v8f   __attribute__((ext_vector_type(8)));

constexpr int NL = 21;          // labels
constexpr int NH = 96, NW = 96;
constexpr int NP = NH * NW;     // 9216 pixels
constexpr int MPAD = 32;        // padded label rows for WMMA (2 x 16)
constexpr int KCHUNK = 8;       // split-K chunks
constexpr float LOG2E = 1.44269504088896340736f;

// ---------------------------------------------------------------------------
// features: f = (y/70, x/70, r/12, g/12, b/12)
__global__ void k_feats(const float* __restrict__ ref, float* __restrict__ feats) {
    int p = blockIdx.x * blockDim.x + threadIdx.x;
    if (p >= NP) return;
    float y = (float)(p / NW), x = (float)(p % NW);
    feats[0 * NP + p] = y * (1.0f / 70.0f);
    feats[1 * NP + p] = x * (1.0f / 70.0f);
    feats[2 * NP + p] = ref[0 * NP + p] * (1.0f / 12.0f);
    feats[3 * NP + p] = ref[1 * NP + p] * (1.0f / 12.0f);
    feats[4 * NP + p] = ref[2 * NP + p] * (1.0f / 12.0f);
}

// ---------------------------------------------------------------------------
// One block per row i: K[i][j] = exp(-0.5*d2) stored bf16, plus row sum.
__global__ void __launch_bounds__(256)
k_buildK(const float* __restrict__ feats, bf16* __restrict__ Kmat,
         float* __restrict__ normsum) {
    const int i = blockIdx.x;
    const float fi0 = feats[0 * NP + i], fi1 = feats[1 * NP + i],
                fi2 = feats[2 * NP + i], fi3 = feats[3 * NP + i],
                fi4 = feats[4 * NP + i];
    float s = 0.0f;
    bf16* __restrict__ krow = Kmat + (size_t)i * NP;
    for (int j = threadIdx.x; j < NP; j += 256) {
        float d0 = fi0 - feats[0 * NP + j];
        float d1 = fi1 - feats[1 * NP + j];
        float d2_ = fi2 - feats[2 * NP + j];
        float d3 = fi3 - feats[3 * NP + j];
        float d4 = fi4 - feats[4 * NP + j];
        float dd = d0 * d0 + d1 * d1 + d2_ * d2_ + d3 * d3 + d4 * d4;
        float kv = exp2f(-0.5f * LOG2E * dd);   // exp(-0.5*dd)
        krow[j] = (bf16)kv;
        s += kv;
    }
    __shared__ float red[256];
    red[threadIdx.x] = s;
    __syncthreads();
    for (int off = 128; off > 0; off >>= 1) {
        if (threadIdx.x < off) red[threadIdx.x] += red[threadIdx.x + off];
        __syncthreads();
    }
    if (threadIdx.x == 0) normsum[i] = red[0];
}

// ---------------------------------------------------------------------------
// rnorm = 1/(sqrt(rowsum)+1e-8); also build normalized 71-tap 1D Gaussian.
__global__ void k_norm(const float* __restrict__ normsum, float* __restrict__ rnorm,
                       float* __restrict__ w71) {
    int t = blockIdx.x * blockDim.x + threadIdx.x;
    if (t < NP) rnorm[t] = 1.0f / (sqrtf(normsum[t]) + 1e-8f);
    if (t < 71) {
        float s = 0.0f;
        for (int d = 0; d < 71; ++d) {
            float dd = (float)(d - 35);
            s += exp2f(-dd * dd * (LOG2E / 72.0f));
        }
        float dd = (float)(t - 35);
        w71[t] = exp2f(-dd * dd * (LOG2E / 72.0f)) / s;
    }
}

// ---------------------------------------------------------------------------
// U = log(clip(u,1e-5,1)); q = softmax(U) over labels.
__global__ void k_initU(const float* __restrict__ u, float* __restrict__ U,
                        float* __restrict__ q) {
    int p = blockIdx.x * blockDim.x + threadIdx.x;
    if (p >= NP) return;
    float h[NL];
    float mx = -1e30f;
#pragma unroll
    for (int l = 0; l < NL; ++l) {
        float v = u[l * NP + p];
        v = fminf(fmaxf(v, 1e-5f), 1.0f);
        v = logf(v);
        U[l * NP + p] = v;
        h[l] = v;
        mx = fmaxf(mx, v);
    }
    float ssum = 0.0f;
#pragma unroll
    for (int l = 0; l < NL; ++l) {
        float e = exp2f((h[l] - mx) * LOG2E);
        h[l] = e;
        ssum += e;
    }
    float inv = 1.0f / ssum;
#pragma unroll
    for (int l = 0; l < NL; ++l) q[l * NP + p] = h[l] * inv;
}

// ---------------------------------------------------------------------------
// A[l][p] = bf16(q[l][p]*rnorm[p]) for l<21, zeros for padded rows.
__global__ void k_prepA(const float* __restrict__ q, const float* __restrict__ rnorm,
                        bf16* __restrict__ A) {
    int idx = blockIdx.x * blockDim.x + threadIdx.x;
    if (idx >= MPAD * NP) return;
    int l = idx / NP;
    int p = idx - l * NP;
    A[idx] = (l < NL) ? (bf16)(q[idx] * rnorm[p]) : (bf16)0.0f;
}

// ---------------------------------------------------------------------------
// Load one 16x32 bf16 tile operand in the CDNA5 WMMA lane layout.
// lane r=lane&15 -> matrix row; hi half-wave takes K-offsets +8 / +24.
// Two contiguous 16-byte loads per lane.
__device__ __forceinline__ v16bf ldtile(const bf16* __restrict__ p) {
    v8bf lo = *reinterpret_cast<const v8bf*>(p);
    v8bf hi = *reinterpret_cast<const v8bf*>(p + 16);
    return __builtin_shufflevector(lo, hi, 0, 1, 2, 3, 4, 5, 6, 7,
                                   8, 9, 10, 11, 12, 13, 14, 15);
}

// ---------------------------------------------------------------------------
// C[l][n] = sum_k A[l][k] * K[k][n]   (K symmetric -> B tiles read as K rows)
// grid = (NP/256 n-tiles, KCHUNK k-chunks), 256 threads = 8 waves.
// Each wave: 32 output cols, both 16-row M tiles -> 4 WMMA accumulators.
// WMMA order c00->c01->c11->c10 chains operand-reuse hints (RA/RB).
__global__ void __launch_bounds__(256)
k_gemm(const bf16* __restrict__ A, const bf16* __restrict__ Kmat,
       float* __restrict__ Cpart) {
    const int lane = threadIdx.x & 31;
    const int wave = threadIdx.x >> 5;
    const int r    = lane & 15;
    const int hi   = lane >> 4;
    const int base = hi * 8;
    const int n0   = blockIdx.x * 256 + wave * 32;
    const int kc   = blockIdx.y;
    const int kbeg = kc * (NP / KCHUNK);
    const int kend = kbeg + (NP / KCHUNK);

    v8f c00 = {}, c01 = {}, c10 = {}, c11 = {};

    const bf16* __restrict__ arow  = A + (size_t)r * NP + base;
    const bf16* __restrict__ arow2 = arow + (size_t)16 * NP;
    const bf16* __restrict__ b0row = Kmat + (size_t)(n0 + r) * NP + base;
    const bf16* __restrict__ b1row = b0row + (size_t)16 * NP;

#pragma unroll 2
    for (int k0 = kbeg; k0 < kend; k0 += 32) {
        v16bf a0 = ldtile(arow + k0);
        v16bf a1 = ldtile(arow2 + k0);
        v16bf b0 = ldtile(b0row + k0);
        v16bf b1 = ldtile(b1row + k0);
        // prefetch next k-step of the streaming (K) operand
        __builtin_prefetch(b0row + k0 + 64, 0, 1);
        __builtin_prefetch(b1row + k0 + 64, 0, 1);
        // chain operand reuse: a0 -> a0, b1 -> b1, a1 -> a1
        c00 = __builtin_amdgcn_wmma_f32_16x16x32_bf16(false, a0, false, b0,
                                                      (short)0, c00, true,  false);
        c01 = __builtin_amdgcn_wmma_f32_16x16x32_bf16(false, a0, false, b1,
                                                      (short)0, c01, false, true);
        c11 = __builtin_amdgcn_wmma_f32_16x16x32_bf16(false, a1, false, b1,
                                                      (short)0, c11, true,  false);
        c10 = __builtin_amdgcn_wmma_f32_16x16x32_bf16(false, a1, false, b0,
                                                      (short)0, c10, false, false);
    }

    // 16x16 f32 C/D layout: VGPR e -> M = e + hi*8 ; N = lane&15
    float* __restrict__ Cc = Cpart + (size_t)kc * NL * NP;
    const int n_lo = n0 + r;
    const int n_hi = n0 + 16 + r;
#pragma unroll
    for (int e = 0; e < 8; ++e) {
        int m = e + hi * 8;          // 0..15: always < NL
        Cc[(size_t)m * NP + n_lo] = c00[e];
        Cc[(size_t)m * NP + n_hi] = c01[e];
        int m2 = m + 16;             // 16..31: only rows < 21 written
        if (m2 < NL) {
            Cc[(size_t)m2 * NP + n_lo] = c10[e];
            Cc[(size_t)m2 * NP + n_hi] = c11[e];
        }
    }
}

// ---------------------------------------------------------------------------
// Separable spatial Gaussian: horizontal pass (one block per (l,y) row).
__global__ void k_convh(const float* __restrict__ q, const float* __restrict__ w,
                        float* __restrict__ tmp) {
    const int row = blockIdx.x;          // l*96 + y
    const int x = threadIdx.x;
    __shared__ float wsh[71];
    if (threadIdx.x < 71) wsh[threadIdx.x] = w[threadIdx.x];
    __syncthreads();
    if (x >= NW) return;
    const float* __restrict__ src = q + (size_t)row * NW;
    float s = 0.0f;
    for (int d = 0; d < 71; ++d) {
        int xx = x + d - 35;
        if (xx >= 0 && xx < NW) s += wsh[d] * src[xx];
    }
    tmp[(size_t)row * NW + x] = s;
}

// vertical pass (one thread per element).
__global__ void k_convv(const float* __restrict__ tmp, const float* __restrict__ w,
                        float* __restrict__ qsf) {
    int idx = blockIdx.x * blockDim.x + threadIdx.x;
    if (idx >= NL * NP) return;
    int l = idx / NP;
    int rem = idx - l * NP;
    int y = rem / NW;
    int x = rem - y * NW;
    const float* __restrict__ src = tmp + (size_t)l * NP;
    float s = 0.0f;
    for (int d = 0; d < 71; ++d) {
        int yy = y + d - 35;
        if (yy >= 0 && yy < NH) s += w[d] * src[yy * NW + x];
    }
    qsf[idx] = s;
}

// ---------------------------------------------------------------------------
// q = softmax(U + 4*(sum_k Cpart)*rnorm + 2*qsf) over labels.
__global__ void k_combine(const float* __restrict__ U, const float* __restrict__ Cpart,
                          const float* __restrict__ rnorm, const float* __restrict__ qsf,
                          float* __restrict__ q) {
    int p = blockIdx.x * blockDim.x + threadIdx.x;
    if (p >= NP) return;
    const float rn = rnorm[p];
    float h[NL];
    float mx = -1e30f;
#pragma unroll
    for (int l = 0; l < NL; ++l) {
        float cb = 0.0f;
#pragma unroll
        for (int c = 0; c < KCHUNK; ++c)
            cb += Cpart[((size_t)c * NL + l) * NP + p];
        float v = U[l * NP + p] + 4.0f * cb * rn + 2.0f * qsf[l * NP + p];
        h[l] = v;
        mx = fmaxf(mx, v);
    }
    float ssum = 0.0f;
#pragma unroll
    for (int l = 0; l < NL; ++l) {
        float e = exp2f((h[l] - mx) * LOG2E);
        h[l] = e;
        ssum += e;
    }
    float inv = 1.0f / ssum;
#pragma unroll
    for (int l = 0; l < NL; ++l) q[l * NP + p] = h[l] * inv;
}

// ---------------------------------------------------------------------------
extern "C" void kernel_launch(void* const* d_in, const int* in_sizes, int n_in,
                              void* d_out, int out_size, void* d_ws, size_t ws_size,
                              hipStream_t stream) {
    const float* u   = (const float*)d_in[0];   // [1,21,96,96]
    const float* ref = (const float*)d_in[1];   // [1,3,96,96]

    // workspace carving (256B aligned); total ~181 MB, dominated by bf16 K.
    char* wp = (char*)d_ws;
    auto carve = [&](size_t bytes) -> char* {
        char* r = wp;
        wp += (bytes + 255) & ~(size_t)255;
        return r;
    };
    bf16*  Kmat    = (bf16*)carve((size_t)NP * NP * sizeof(bf16));     // 170 MB
    bf16*  Abf     = (bf16*)carve((size_t)MPAD * NP * sizeof(bf16));
    float* feats   = (float*)carve((size_t)5 * NP * sizeof(float));
    float* normsum = (float*)carve((size_t)NP * sizeof(float));
    float* rnorm   = (float*)carve((size_t)NP * sizeof(float));
    float* w71     = (float*)carve(71 * sizeof(float));
    float* Ubuf    = (float*)carve((size_t)NL * NP * sizeof(float));
    float* qbuf    = (float*)carve((size_t)NL * NP * sizeof(float));
    float* Cpart   = (float*)carve((size_t)KCHUNK * NL * NP * sizeof(float));
    float* tmp     = (float*)carve((size_t)NL * NP * sizeof(float));
    float* qsf     = (float*)carve((size_t)NL * NP * sizeof(float));

    // one-time setup
    k_feats<<<(NP + 255) / 256, 256, 0, stream>>>(ref, feats);
    k_buildK<<<NP, 256, 0, stream>>>(feats, Kmat, normsum);
    k_norm<<<(NP + 255) / 256, 256, 0, stream>>>(normsum, rnorm, w71);
    k_initU<<<(NP + 127) / 128, 128, 0, stream>>>(u, Ubuf, qbuf);

    for (int it = 0; it < 5; ++it) {
        k_prepA<<<(MPAD * NP + 255) / 256, 256, 0, stream>>>(qbuf, rnorm, Abf);
        k_gemm<<<dim3(NP / 256, KCHUNK), 256, 0, stream>>>(Abf, Kmat, Cpart);
        k_convh<<<NL * NH, 96, 0, stream>>>(qbuf, w71, tmp);
        k_convv<<<(NL * NP + 255) / 256, 256, 0, stream>>>(tmp, w71, qsf);
        k_combine<<<(NP + 127) / 128, 128, 0, stream>>>(Ubuf, Cpart, rnorm, qsf, qbuf);
    }

    hipMemcpyAsync(d_out, qbuf, (size_t)NL * NP * sizeof(float),
                   hipMemcpyDeviceToDevice, stream);
}
